// PromptGetter_60593398612122
// MI455X (gfx1250) — compile-verified, matrix operand order checked by reading, not run on hardware
//
#include <hip/hip_runtime.h>
#include <hip/hip_bf16.h>

typedef __attribute__((ext_vector_type(16))) _Float16 v16h;
typedef __attribute__((ext_vector_type(8)))  float    v8f;

#define THRESH 0.65f
#define NEGV  -1e9f

// ---- workspace layout (bytes), all offsets 256B aligned ----
#define OFF_INVN   0u                       // 4096 f32          (16 KB)
#define OFF_AHI    16384u                   // 4*8*32*8 u32      (32 KB)
#define OFF_ALO    49152u                   // 32 KB
#define OFF_SIM    81920u                   // 64*4096 f32       (1 MB)
#define OFF_CMAX   1130496u                 // 64*256 f32        (64 KB)
#define OFF_CARG   1196032u                 // 64*256 i32
#define OFF_CMIN   1261568u                 // 64*256 f32
#define OFF_CMINA  1327104u                 // 64*256 i32
#define OFF_BHI    1392640u                 // 256*8*32*8 u32    (2 MB)
#define OFF_BLO    3489792u                 // 2 MB  -> total ~5.33 MB

__device__ __forceinline__ unsigned short h2u(_Float16 h) {
  union { _Float16 h; unsigned short u; } x; x.h = h; return x.u;
}

__device__ __forceinline__ v16h load16h(const unsigned int* __restrict__ p) {
  union { unsigned int u[8]; v16h h; } x;
#pragma unroll
  for (int i = 0; i < 8; ++i) x.u[i] = p[i];
  return x.h;
}

// WMMA 16-bit operand lane mapping (ISA 7.12.2): lane gives row (A) / col (B)
// = lane&15; K-halves: lanes<16 -> K base 0, lanes>=16 -> K base 8; dword j:
// j<4 -> K = klo+2j,(+1);  j>=4 -> K = 16+klo+2(j-4),(+1).
__device__ __forceinline__ unsigned wmma_kk(unsigned lane, unsigned j) {
  unsigned klo = (lane >> 4) << 3;
  return (j < 4u) ? (klo + 2u * j) : (16u + klo + 2u * (j - 4u));
}

// ---------- 1) per-pixel inverse L2 norm over 256 channels ----------
__global__ void k_invnorm(const float* __restrict__ emb, float* __restrict__ invn) {
  int p = blockIdx.x * blockDim.x + threadIdx.x;   // 0..4095
  float s = 0.f;
#pragma unroll 8
  for (int c = 0; c < 256; ++c) { float v = emb[c * 4096 + p]; s += v * v; }
  invn[p] = rsqrtf(s);
}

// ---------- 2) pack normalized embeddings as WMMA B (hi/lo f16 split) ----------
__global__ void k_packB(const float* __restrict__ emb, const float* __restrict__ invn,
                        unsigned int* __restrict__ bhi, unsigned int* __restrict__ blo) {
  unsigned t = blockIdx.x * blockDim.x + threadIdx.x;   // 256*8*32*8 = 524288
  unsigned j = t & 7u, lane = (t >> 3) & 31u, kt = (t >> 8) & 7u, nt = t >> 11;
  unsigned n = lane & 15u;
  unsigned k0 = kt * 32u + wmma_kk(lane, j);
  unsigned p = nt * 16u + n;
  float inv = invn[p];
  float v0 = emb[k0 * 4096u + p] * inv;
  float v1 = emb[(k0 + 1u) * 4096u + p] * inv;
  _Float16 h0 = (_Float16)v0, h1 = (_Float16)v1;
  _Float16 l0 = (_Float16)(v0 - (float)h0), l1 = (_Float16)(v1 - (float)h1);
  bhi[t] = ((unsigned)h2u(h1) << 16) | h2u(h0);
  blo[t] = ((unsigned)h2u(l1) << 16) | h2u(l0);
}

// ---------- 3) pack reference feats as WMMA A (hi/lo f16 split) ----------
__global__ void k_packA(const float* __restrict__ refs,
                        unsigned int* __restrict__ ahi, unsigned int* __restrict__ alo) {
  unsigned t = blockIdx.x * blockDim.x + threadIdx.x;   // 4*8*32*8 = 8192
  unsigned j = t & 7u, lane = (t >> 3) & 31u, kt = (t >> 8) & 7u, mt = t >> 11;
  unsigned m = lane & 15u;
  unsigned k0 = kt * 32u + wmma_kk(lane, j);
  const float* row = refs + (mt * 16u + m) * 256u;
  float v0 = row[k0], v1 = row[k0 + 1u];
  _Float16 h0 = (_Float16)v0, h1 = (_Float16)v1;
  _Float16 l0 = (_Float16)(v0 - (float)h0), l1 = (_Float16)(v1 - (float)h1);
  ahi[t] = ((unsigned)h2u(h1) << 16) | h2u(h0);
  alo[t] = ((unsigned)h2u(l1) << 16) | h2u(l0);
}

// ---------- 4) GEMM: sim(64x4096) = refs(64x256) x embN(256x4096) ----------
// One wave per 16x16 tile; fp32 accuracy via f16 hi/lo split: 3 WMMAs / K-tile.
__global__ void __launch_bounds__(32) k_gemm(const unsigned int* __restrict__ ahi,
                                             const unsigned int* __restrict__ alo,
                                             const unsigned int* __restrict__ bhi,
                                             const unsigned int* __restrict__ blo,
                                             float* __restrict__ sim) {
  const int lane = threadIdx.x;
  const int nt = blockIdx.x;   // 0..255 (N tiles of 16 pixels)
  const int mt = blockIdx.y;   // 0..3   (M tiles of 16 layers)
  v8f acc = {};
#pragma unroll
  for (int kt = 0; kt < 8; ++kt) {
    const unsigned aoff = (((unsigned)(mt * 8 + kt) * 32u) + lane) * 8u;
    const unsigned boff = (((unsigned)(nt * 8 + kt) * 32u) + lane) * 8u;
    v16h ah = load16h(ahi + aoff);
    v16h al = load16h(alo + aoff);
    v16h bh = load16h(bhi + boff);
    v16h bl = load16h(blo + boff);
    acc = __builtin_amdgcn_wmma_f32_16x16x32_f16(false, al, false, bh, (short)0, acc, false, false);
    acc = __builtin_amdgcn_wmma_f32_16x16x32_f16(false, ah, false, bl, (short)0, acc, false, false);
    acc = __builtin_amdgcn_wmma_f32_16x16x32_f16(false, ah, false, bh, (short)0, acc, false, false);
  }
  // C/D layout: VGPR r -> M = r (+8 for lanes 16-31), N = lane&15
  const int n  = lane & 15;
  const int mh = (lane >> 4) << 3;
#pragma unroll
  for (int r = 0; r < 8; ++r)
    sim[(mt * 16 + r + mh) * 4096 + nt * 16 + n] = acc[r];
}

// ---------- 5) per-(layer, grid-cell) scan: on-the-fly bilinear upsample ----------
// Never materializes the 1024x1024 map: each 64x64 output block depends only on
// a 6x6 patch of the 64x64 sim map (half-pixel centers, scale 16, edge clamp).
__global__ void __launch_bounds__(256) k_scan(const float* __restrict__ sim,
                                              float* __restrict__ cmax, int* __restrict__ carg,
                                              float* __restrict__ cmin, int* __restrict__ cmina) {
  __shared__ float patch[36];
  __shared__ float redv[256];
  __shared__ int   redi[256];
  const int cell = blockIdx.x;        // 0..255
  const int l    = blockIdx.y;        // 0..63
  const int t    = threadIdx.x;
  const int gy = cell >> 4, gx = cell & 15;
  const float* simL = sim + l * 4096;
  if (t < 36) {
    int i = t / 6, jj = t % 6;
    int ry = min(max(gy * 4 - 1 + i, 0), 63);
    int rx = min(max(gx * 4 - 1 + jj, 0), 63);
    patch[t] = simL[ry * 64 + rx];
  }
  __syncthreads();

  float tmax = -3e38f; int tmaxi = 0;
  float tmin =  3e38f; int tmini = 0;
#pragma unroll
  for (int q = 0; q < 16; ++q) {
    int pid = t * 16 + q;              // in-block flat index (row-major, matches jnp)
    int oy = pid >> 6, ox = pid & 63;
    int py = (oy + 8) >> 4;            // local src row in patch (0..4)
    int px = (ox + 8) >> 4;
    float fy = (oy + 8.5f) * 0.0625f - (float)py;
    float fx = (ox + 8.5f) * 0.0625f - (float)px;
    float p00 = patch[py * 6 + px],     p01 = patch[py * 6 + px + 1];
    float p10 = patch[py * 6 + px + 6], p11 = patch[py * 6 + px + 7];
    float v = (1.f - fy) * ((1.f - fx) * p00 + fx * p01)
            +        fy  * ((1.f - fx) * p10 + fx * p11);
    float mv = (v > THRESH) ? v : NEGV;
    if (mv > tmax) { tmax = mv; tmaxi = pid; }
    int gidx = (gy * 64 + oy) * 1024 + (gx * 64 + ox);  // full-image flat index
    if (v < tmin || (v == tmin && gidx < tmini)) { tmin = v; tmini = gidx; }
  }

  // reduce max (tie -> smaller in-block index; per-thread pids are ordered)
  redv[t] = tmax; redi[t] = tmaxi;
  for (int s = 128; s > 0; s >>= 1) {
    __syncthreads();
    if (t < s) {
      float v2 = redv[t + s]; int i2 = redi[t + s];
      if (v2 > redv[t] || (v2 == redv[t] && i2 < redi[t])) { redv[t] = v2; redi[t] = i2; }
    }
  }
  if (t == 0) { cmax[l * 256 + cell] = redv[0]; carg[l * 256 + cell] = redi[0]; }
  __syncthreads();

  // reduce min (tie -> smaller global flat index)
  redv[t] = tmin; redi[t] = tmini;
  for (int s = 128; s > 0; s >>= 1) {
    __syncthreads();
    if (t < s) {
      float v2 = redv[t + s]; int i2 = redi[t + s];
      if (v2 < redv[t] || (v2 == redv[t] && i2 < redi[t])) { redv[t] = v2; redi[t] = i2; }
    }
  }
  if (t == 0) { cmin[l * 256 + cell] = redv[0]; cmina[l * 256 + cell] = redi[0]; }
}

// ---------- 6) per-layer: stable sort points by score desc; bg = global argmin ----------
__global__ void __launch_bounds__(256) k_final(const float* __restrict__ cmax,
                                               const int* __restrict__ carg,
                                               const float* __restrict__ cmin,
                                               const int* __restrict__ cmina,
                                               float* __restrict__ out) {
  __shared__ unsigned long long keys[256];
  __shared__ float redv[256];
  __shared__ int   redi[256];
  const int l = blockIdx.x, t = threadIdx.x;

  float best = cmax[l * 256 + t];
  bool valid = best > (NEGV * 0.5f);
  float score = valid ? best : 0.0f;                 // matches where(valid, best, 0)
  unsigned sb = __float_as_uint(score);              // score >= 0 -> bit order monotonic
  keys[t] = ((unsigned long long)(0xFFFFFFFFu - sb) << 32) | (unsigned)t;
  __syncthreads();

  // bitonic sort ascending: (score desc, cell idx asc) == stable argsort(-score)
  for (int ksz = 2; ksz <= 256; ksz <<= 1) {
    for (int jsz = ksz >> 1; jsz > 0; jsz >>= 1) {
      __syncthreads();
      int ixj = t ^ jsz;
      if (ixj > t) {
        bool up = ((t & ksz) == 0);
        unsigned long long a = keys[t], b = keys[ixj];
        if ((a > b) == up) { keys[t] = b; keys[ixj] = a; }
      }
    }
  }
  __syncthreads();

  int cell = (int)(keys[t] & 0xFFu);
  float b2 = cmax[l * 256 + cell];
  int  idx = carg[l * 256 + cell];
  bool v2  = b2 > (NEGV * 0.5f);
  int gy = cell >> 4, gx = cell & 15;
  float xo = v2 ? (float)(gx * 64 + (idx & 63)) : 0.f;
  float yo = v2 ? (float)(gy * 64 + (idx >> 6)) : 0.f;
  float so = v2 ? b2 : 0.f;
  float* pt = out + (l * 256 + t) * 3;
  pt[0] = xo; pt[1] = yo; pt[2] = so;

  // background: min over all 256 cell minima (tie -> smaller flat index)
  redv[t] = cmin[l * 256 + t]; redi[t] = cmina[l * 256 + t];
  for (int s = 128; s > 0; s >>= 1) {
    __syncthreads();
    if (t < s) {
      float v3 = redv[t + s]; int i3 = redi[t + s];
      if (v3 < redv[t] || (v3 == redv[t] && i3 < redi[t])) { redv[t] = v3; redi[t] = i3; }
    }
  }
  if (t == 0) {
    int a = redi[0];
    float* bg = out + 64 * 256 * 3 + l * 2;
    bg[0] = (float)(a & 1023);   // x = col
    bg[1] = (float)(a >> 10);    // y = row
  }
}

extern "C" void kernel_launch(void* const* d_in, const int* in_sizes, int n_in,
                              void* d_out, int out_size, void* d_ws, size_t ws_size,
                              hipStream_t stream) {
  const float* emb  = (const float*)d_in[0];   // (1,256,64,64)
  const float* refs = (const float*)d_in[1];   // (64,1,256)
  (void)d_in; (void)in_sizes; (void)n_in; (void)out_size; (void)ws_size;

  char* ws = (char*)d_ws;
  float*        invn  = (float*)(ws + OFF_INVN);
  unsigned int* ahi   = (unsigned int*)(ws + OFF_AHI);
  unsigned int* alo   = (unsigned int*)(ws + OFF_ALO);
  float*        sim   = (float*)(ws + OFF_SIM);
  float*        cmax  = (float*)(ws + OFF_CMAX);
  int*          carg  = (int*)(ws + OFF_CARG);
  float*        cmin  = (float*)(ws + OFF_CMIN);
  int*          cmina = (int*)(ws + OFF_CMINA);
  unsigned int* bhi   = (unsigned int*)(ws + OFF_BHI);
  unsigned int* blo   = (unsigned int*)(ws + OFF_BLO);
  float* out = (float*)d_out;

  k_invnorm<<<16, 256, 0, stream>>>(emb, invn);
  k_packB<<<2048, 256, 0, stream>>>(emb, invn, bhi, blo);
  k_packA<<<32, 256, 0, stream>>>(refs, ahi, alo);
  k_gemm<<<dim3(256, 4), 32, 0, stream>>>(ahi, alo, bhi, blo, sim);
  k_scan<<<dim3(256, 64), 256, 0, stream>>>(sim, cmax, carg, cmin, cmina);
  k_final<<<64, 256, 0, stream>>>(cmax, carg, cmin, cmina, out);
}
